// MoELayer_8366596293045
// MI455X (gfx1250) — matrix operand order repaired and checked by reading, not compiled, not run on hardware
//
#include <hip/hip_runtime.h>

// ---------------------------------------------------------------------------
// MoE (LLaMA SwiGLU experts, top-2 routing + 2 shared experts) for gfx1250.
// bf16 WMMA (v_wmma_f32_16x16x32_bf16) with fp32 accumulation.
// Double-buffered LDS staging, one barrier per K-step, prefetch on weights,
// packed f32->bf16 conversion (v_cvt_pk_bf16_f32) via __builtin_convertvector.
// ---------------------------------------------------------------------------

constexpr int T    = 4096;   // B*S tokens
constexpr int DIM  = 1024;
constexpr int HID  = 2048;
constexpr int E    = 8;
constexpr int NSH  = 2;
constexpr int KT   = 32;     // K step per WMMA (bf16 16x16x32)
constexpr int LDA  = 40;     // LDS row stride (elements) for A tiles, padded
constexpr int LDB  = 40;     // LDS row stride (elements) for B tiles, padded

typedef __attribute__((ext_vector_type(16))) __bf16        bf16x16;
typedef __attribute__((ext_vector_type(2)))  __bf16        bf16x2;
typedef __attribute__((ext_vector_type(8)))  float         f32x8;
typedef __attribute__((ext_vector_type(4)))  float         f32x4;
typedef __attribute__((ext_vector_type(2)))  float         f32x2;
typedef __attribute__((ext_vector_type(4)))  unsigned int  u32x4;
typedef __attribute__((ext_vector_type(2)))  unsigned int  u32x2;

__device__ __forceinline__ unsigned int pack2bf(float a, float b) {
    f32x2 v; v.x = a; v.y = b;
    bf16x2 r = __builtin_convertvector(v, bf16x2);   // v_cvt_pk_bf16_f32 (RNE)
    return __builtin_bit_cast(unsigned int, r);
}
__device__ __forceinline__ unsigned short bf16c(float f) {
    return __builtin_bit_cast(unsigned short, (__bf16)f);
}

union AFrag { u32x4 q[2]; bf16x16 v; };

// ---------------------------------------------------------------------------
// Router: one wave (32 lanes) per token. Computes 8 logits, top-2 softmax,
// appends (token, weight, h-slot) into per-expert gather lists via atomics.
// ---------------------------------------------------------------------------
__global__ __launch_bounds__(256)
void router_kernel(const float* __restrict__ x, const float* __restrict__ rw,
                   const float* __restrict__ eb, int* __restrict__ cnt,
                   int* __restrict__ idx, float* __restrict__ wt,
                   int* __restrict__ slot) {
    const int wave = threadIdx.x >> 5;
    const int lane = threadIdx.x & 31;
    const int t = blockIdx.x * 8 + wave;
    if (t >= T) return;

    float acc[E];
#pragma unroll
    for (int e = 0; e < E; ++e) acc[e] = 0.0f;

    const float* xr = x + (size_t)t * DIM;
    for (int d = lane; d < DIM; d += 32) {
        const float xv = xr[d];
#pragma unroll
        for (int e = 0; e < E; ++e) acc[e] += xv * rw[e * DIM + d];
    }
#pragma unroll
    for (int off = 16; off > 0; off >>= 1) {
#pragma unroll
        for (int e = 0; e < E; ++e) acc[e] += __shfl_xor(acc[e], off, 32);
    }

    if (lane == 0) {
#pragma unroll
        for (int e = 0; e < E; ++e) acc[e] += eb[e];
        int i0 = 0;
#pragma unroll
        for (int e = 1; e < E; ++e) if (acc[e] > acc[i0]) i0 = e;
        int i1 = (i0 == 0) ? 1 : 0;
#pragma unroll
        for (int e = 0; e < E; ++e)
            if (e != i0 && acc[e] > acc[i1]) i1 = e;

        const float e1 = __expf(acc[i1] - acc[i0]);
        const float inv = __builtin_amdgcn_rcpf(1.0f + e1);
        const float w0 = inv, w1v = e1 * inv;

        int p0 = atomicAdd(&cnt[i0], 1);
        idx[i0 * T + p0] = t; wt[i0 * T + p0] = w0;  slot[i0 * T + p0] = t * 2;
        int p1 = atomicAdd(&cnt[i1], 1);
        idx[i1 * T + p1] = t; wt[i1 * T + p1] = w1v; slot[i1 * T + p1] = t * 2 + 1;
    }
}

// ---------------------------------------------------------------------------
// Cast x fp32 -> bf16 (A-matrix, reused by all GEMMs).
// ---------------------------------------------------------------------------
__global__ __launch_bounds__(256)
void cast_kernel(const float* __restrict__ x, unsigned short* __restrict__ xbf) {
    const int i = (blockIdx.x * 256 + threadIdx.x) * 4;
    if (i < T * DIM) {
        f32x4 v = *(const f32x4*)(x + i);
        u32x2 o; o.x = pack2bf(v.x, v.y); o.y = pack2bf(v.z, v.w);
        *(u32x2*)(xbf + i) = o;
    }
}

// ---------------------------------------------------------------------------
// Up-projection + SwiGLU: h[rows,HID] = silu(Xg @ W1^T) * (Xg @ W3^T), bf16.
// ident_rows > 0 => identity gather over ident_rows rows per z-slice (shared
// experts; h rows at e*ident_rows + r), else gather via per-expert lists.
// Block: 256 thr = 8 waves; tile 64(M) x 128(N); wave tile 32x32.
// Double-buffered LDS, single barrier per K-step; next-tile global loads
// issued before the WMMA burst so HBM/L2 latency overlaps compute.
// ---------------------------------------------------------------------------
__global__ __launch_bounds__(256)
void ffn_up_kernel(const unsigned short* __restrict__ xbf,
                   const float* __restrict__ w1all,
                   const float* __restrict__ w3all,
                   const int* __restrict__ cnt, const int* __restrict__ idx,
                   const int* __restrict__ slot,
                   unsigned short* __restrict__ hout, int ident_rows) {
    __shared__ __align__(16) unsigned short sA [2][64 * LDA];
    __shared__ __align__(16) unsigned short sB1[2][128 * LDB];
    __shared__ __align__(16) unsigned short sB3[2][128 * LDB];

    const int e    = blockIdx.z;
    const int rows = (ident_rows > 0) ? ident_rows : cnt[e];
    const int m0   = blockIdx.y * 64;
    if (m0 >= rows) return;
    const int n0   = blockIdx.x * 128;

    const float* W1 = w1all + (size_t)e * HID * DIM;
    const float* W3 = w3all + (size_t)e * HID * DIM;

    const int tid = threadIdx.x;
    // A staging: thread -> (row = tid/4, 8-elem segment = tid%4)
    const int ra = tid >> 2, aseg = tid & 3;
    int pa = m0 + ra; if (pa >= rows) pa = rows - 1;
    const int ta = (ident_rows > 0) ? pa : idx[e * T + pa];
    const unsigned short* agp = xbf + (size_t)ta * DIM + aseg * 8;

    // B staging: thread -> (row = tid/2 of weight tile, 16-float half = tid%2)
    const int rb = tid >> 1, bhalf = tid & 1;
    const float* b1p = W1 + (size_t)(n0 + rb) * DIM + bhalf * 16;
    const float* b3p = W3 + (size_t)(n0 + rb) * DIM + bhalf * 16;

    const int wave = tid >> 5, lane = tid & 31;
    const int wm = wave & 1, wn = wave >> 1;      // 2 x 4 wave grid
    const int lr  = lane & 15;
    const int aks = (lane < 16) ? 0 : 16;         // A: K {0..15} vs {16..31}
    const int bko = (lane < 16) ? 0 : 16;         // B: K {0..15} vs {16..31}

    f32x8 G[2][2] = {};
    f32x8 U[2][2] = {};

    // prologue: fetch K-tile 0
    u32x4 av = *(const u32x4*)(agp);
    f32x4 b1v[4], b3v[4];
#pragma unroll
    for (int j = 0; j < 4; ++j) {
        b1v[j] = *(const f32x4*)(b1p + j * 4);
        b3v[j] = *(const f32x4*)(b3p + j * 4);
    }

    int p = 0;
    for (int k0 = 0; k0 < DIM; k0 += KT, p ^= 1) {
        // stage tile k0 into buffer p (safe: readers of p drained before the
        // previous iteration's barrier)
        *(u32x4*)(sA[p] + ra * LDA + aseg * 8) = av;
        {
            u32x4 o;
            o.x = pack2bf(b1v[0].x, b1v[0].y); o.y = pack2bf(b1v[0].z, b1v[0].w);
            o.z = pack2bf(b1v[1].x, b1v[1].y); o.w = pack2bf(b1v[1].z, b1v[1].w);
            *(u32x4*)(sB1[p] + rb * LDB + bhalf * 16) = o;
            o.x = pack2bf(b1v[2].x, b1v[2].y); o.y = pack2bf(b1v[2].z, b1v[2].w);
            o.z = pack2bf(b1v[3].x, b1v[3].y); o.w = pack2bf(b1v[3].z, b1v[3].w);
            *(u32x4*)(sB1[p] + rb * LDB + bhalf * 16 + 8) = o;
            o.x = pack2bf(b3v[0].x, b3v[0].y); o.y = pack2bf(b3v[0].z, b3v[0].w);
            o.z = pack2bf(b3v[1].x, b3v[1].y); o.w = pack2bf(b3v[1].z, b3v[1].w);
            *(u32x4*)(sB3[p] + rb * LDB + bhalf * 16) = o;
            o.x = pack2bf(b3v[2].x, b3v[2].y); o.y = pack2bf(b3v[2].z, b3v[2].w);
            o.z = pack2bf(b3v[3].x, b3v[3].y); o.w = pack2bf(b3v[3].z, b3v[3].w);
            *(u32x4*)(sB3[p] + rb * LDB + bhalf * 16 + 8) = o;
        }
        __syncthreads();

        // issue next tile's global loads; they retire while WMMAs run
        if (k0 + KT < DIM) {
            const int kn = k0 + KT;
            av = *(const u32x4*)(agp + kn);
#pragma unroll
            for (int j = 0; j < 4; ++j) {
                b1v[j] = *(const f32x4*)(b1p + kn + j * 4);
                b3v[j] = *(const f32x4*)(b3p + kn + j * 4);
            }
            __builtin_prefetch(b1p + kn + KT, 0, 0);  // global_prefetch_b8
            __builtin_prefetch(b3p + kn + KT, 0, 0);
        }

        AFrag a[2], b1f[2], b3f[2];
#pragma unroll
        for (int mi = 0; mi < 2; ++mi) {
            const unsigned short* ap = sA[p] + (wm * 32 + mi * 16 + lr) * LDA;
            a[mi].q[0] = *(const u32x4*)(ap + aks);
            a[mi].q[1] = *(const u32x4*)(ap + aks + 8);
        }
#pragma unroll
        for (int nj = 0; nj < 2; ++nj) {
            const unsigned short* bp1 = sB1[p] + (wn * 32 + nj * 16 + lr) * LDB + bko;
            b1f[nj].q[0] = *(const u32x4*)(bp1);
            b1f[nj].q[1] = *(const u32x4*)(bp1 + 8);
            const unsigned short* bp3 = sB3[p] + (wn * 32 + nj * 16 + lr) * LDB + bko;
            b3f[nj].q[0] = *(const u32x4*)(bp3);
            b3f[nj].q[1] = *(const u32x4*)(bp3 + 8);
        }
        // nj-outer / mi-inner: both A fragments are operands of the first two
        // WMMAs, keeping them in distinct registers (no WAR reload stall).
#pragma unroll
        for (int nj = 0; nj < 2; ++nj)
#pragma unroll
            for (int mi = 0; mi < 2; ++mi) {
                G[mi][nj] = __builtin_amdgcn_wmma_f32_16x16x32_bf16(
                    false, a[mi].v, false, b1f[nj].v, (short)0, G[mi][nj], false, false);
                U[mi][nj] = __builtin_amdgcn_wmma_f32_16x16x32_bf16(
                    false, a[mi].v, false, b3f[nj].v, (short)0, U[mi][nj], false, false);
            }
    }

    // Epilogue: h = silu(g)*u -> bf16, scattered by h-slot.
    const int mrow_add = (lane < 16) ? 0 : 8;
    const int ncol = lane & 15;
#pragma unroll
    for (int mi = 0; mi < 2; ++mi)
#pragma unroll
        for (int nj = 0; nj < 2; ++nj)
#pragma unroll
            for (int i = 0; i < 8; ++i) {
                const int pr = m0 + wm * 32 + mi * 16 + i + mrow_add;
                if (pr < rows) {
                    const int srow = (ident_rows > 0) ? (e * ident_rows + pr)
                                                      : slot[e * T + pr];
                    const int hcol = n0 + wn * 32 + nj * 16 + ncol;
                    const float g = G[mi][nj][i];
                    const float u = U[mi][nj][i];
                    const float s = g * __builtin_amdgcn_rcpf(1.0f + __expf(-g));
                    hout[(size_t)srow * HID + hcol] = bf16c(s * u);
                }
            }
}

// ---------------------------------------------------------------------------
// Down-projection: out[t] += wt * (Hg @ W2^T), scatter via atomicAdd f32.
// out is zero-initialized by the launcher.
// ---------------------------------------------------------------------------
__global__ __launch_bounds__(256)
void ffn_down_kernel(const unsigned short* __restrict__ hin,
                     const float* __restrict__ w2all,
                     const int* __restrict__ cnt, const int* __restrict__ idx,
                     const float* __restrict__ wt, const int* __restrict__ slot,
                     float* __restrict__ out, int ident_rows) {
    __shared__ __align__(16) unsigned short sA[2][64 * LDA];
    __shared__ __align__(16) unsigned short sB[2][128 * LDB];

    const int e    = blockIdx.z;
    const int rows = (ident_rows > 0) ? ident_rows : cnt[e];
    const int m0   = blockIdx.y * 64;
    if (m0 >= rows) return;
    const int n0   = blockIdx.x * 128;

    const float* W2 = w2all + (size_t)e * DIM * HID;

    const int tid = threadIdx.x;
    const int ra = tid >> 2, aseg = tid & 3;
    int pa = m0 + ra; if (pa >= rows) pa = rows - 1;
    const int sa = (ident_rows > 0) ? (e * ident_rows + pa) : slot[e * T + pa];
    const unsigned short* agp = hin + (size_t)sa * HID + aseg * 8;

    const int rb = tid >> 1, bhalf = tid & 1;
    const float* b2p = W2 + (size_t)(n0 + rb) * HID + bhalf * 16;

    const int wave = tid >> 5, lane = tid & 31;
    const int wm = wave & 1, wn = wave >> 1;
    const int lr  = lane & 15;
    const int aks = (lane < 16) ? 0 : 16;
    const int bko = (lane < 16) ? 0 : 16;

    f32x8 C[2][2] = {};

    u32x4 av = *(const u32x4*)(agp);
    f32x4 bv[4];
#pragma unroll
    for (int j = 0; j < 4; ++j) bv[j] = *(const f32x4*)(b2p + j * 4);

    int p = 0;
    for (int k0 = 0; k0 < HID; k0 += KT, p ^= 1) {
        *(u32x4*)(sA[p] + ra * LDA + aseg * 8) = av;
        {
            u32x4 o;
            o.x = pack2bf(bv[0].x, bv[0].y); o.y = pack2bf(bv[0].z, bv[0].w);
            o.z = pack2bf(bv[1].x, bv[1].y); o.w = pack2bf(bv[1].z, bv[1].w);
            *(u32x4*)(sB[p] + rb * LDB + bhalf * 16) = o;
            o.x = pack2bf(bv[2].x, bv[2].y); o.y = pack2bf(bv[2].z, bv[2].w);
            o.z = pack2bf(bv[3].x, bv[3].y); o.w = pack2bf(bv[3].z, bv[3].w);
            *(u32x4*)(sB[p] + rb * LDB + bhalf * 16 + 8) = o;
        }
        __syncthreads();

        if (k0 + KT < HID) {
            const int kn = k0 + KT;
            av = *(const u32x4*)(agp + kn);
#pragma unroll
            for (int j = 0; j < 4; ++j) bv[j] = *(const f32x4*)(b2p + kn + j * 4);
            __builtin_prefetch(b2p + kn + KT, 0, 0);
        }

        AFrag a[2], bf[2];
#pragma unroll
        for (int mi = 0; mi < 2; ++mi) {
            const unsigned short* ap = sA[p] + (wm * 32 + mi * 16 + lr) * LDA;
            a[mi].q[0] = *(const u32x4*)(ap + aks);
            a[mi].q[1] = *(const u32x4*)(ap + aks + 8);
        }
#pragma unroll
        for (int nj = 0; nj < 2; ++nj) {
            const unsigned short* bp = sB[p] + (wn * 32 + nj * 16 + lr) * LDB + bko;
            bf[nj].q[0] = *(const u32x4*)(bp);
            bf[nj].q[1] = *(const u32x4*)(bp + 8);
        }
#pragma unroll
        for (int nj = 0; nj < 2; ++nj)
#pragma unroll
            for (int mi = 0; mi < 2; ++mi)
                C[mi][nj] = __builtin_amdgcn_wmma_f32_16x16x32_bf16(
                    false, a[mi].v, false, bf[nj].v, (short)0, C[mi][nj], false, false);
    }

    const int mrow_add = (lane < 16) ? 0 : 8;
    const int ncol = lane & 15;
#pragma unroll
    for (int mi = 0; mi < 2; ++mi)
#pragma unroll
        for (int nj = 0; nj < 2; ++nj)
#pragma unroll
            for (int i = 0; i < 8; ++i) {
                const int pr = m0 + wm * 32 + mi * 16 + i + mrow_add;
                if (pr < rows) {
                    const int t  = (ident_rows > 0) ? pr : idx[e * T + pr];
                    const float w = (ident_rows > 0) ? 1.0f : wt[e * T + pr];
                    const int col = n0 + wn * 32 + nj * 16 + ncol;
                    atomicAdd(out + (size_t)t * DIM + col, C[mi][nj][i] * w);
                }
            }
}

// ---------------------------------------------------------------------------
// Host launcher. Workspace layout (bytes):
//   [0,256)            : int cnt[E]
//   +256               : int   idx[E*T]
//   +E*T*4             : float wt [E*T]
//   +E*T*4             : int   slot[E*T]
//   +E*T*4             : bf16  xbf[T*DIM]
//   +T*DIM*2           : bf16  h  [2T*HID]  (shared pass aliases it; stream
//                         order serializes shared-down before expert-up)
// ---------------------------------------------------------------------------
extern "C" void kernel_launch(void* const* d_in, const int* in_sizes, int n_in,
                              void* d_out, int out_size, void* d_ws, size_t ws_size,
                              hipStream_t stream) {
    const float* x   = (const float*)d_in[0];
    const float* rw  = (const float*)d_in[1];
    const float* eb  = (const float*)d_in[2];
    const float* w1  = (const float*)d_in[3];
    const float* w3  = (const float*)d_in[4];
    const float* w2  = (const float*)d_in[5];
    const float* sw1 = (const float*)d_in[6];
    const float* sw3 = (const float*)d_in[7];
    const float* sw2 = (const float*)d_in[8];
    float* out = (float*)d_out;

    char* ws = (char*)d_ws;
    int*            cnt  = (int*)ws;
    int*            idx  = (int*)(ws + 256);
    float*          wt   = (float*)(ws + 256 + (size_t)E * T * 4);
    int*            slot = (int*)(ws + 256 + (size_t)2 * E * T * 4);
    unsigned short* xbf  = (unsigned short*)(ws + 256 + (size_t)3 * E * T * 4);
    unsigned short* hbuf = (unsigned short*)(ws + 256 + (size_t)3 * E * T * 4 +
                                             (size_t)T * DIM * 2);

    hipMemsetAsync(cnt, 0, 256, stream);
    hipMemsetAsync(out, 0, (size_t)T * DIM * sizeof(float), stream);
    router_kernel<<<T / 8, 256, 0, stream>>>(x, rw, eb, cnt, idx, wt, slot);
    cast_kernel<<<(T * DIM) / (256 * 4), 256, 0, stream>>>(x, xbf);

    // Shared experts (dense, identity gather), both z-slices in one launch.
    // Shared expert i uses h rows [i*T, (i+1)*T).
    ffn_up_kernel<<<dim3(HID / 128, T / 64, NSH), 256, 0, stream>>>(
        xbf, sw1, sw3, nullptr, nullptr, nullptr, hbuf, T);
    ffn_down_kernel<<<dim3(DIM / 128, T / 64, NSH), 256, 0, stream>>>(
        hbuf, sw2, nullptr, nullptr, nullptr, nullptr, out, T);

    // Routed experts (gathered rows, scatter-add with routing weights).
    ffn_up_kernel<<<dim3(HID / 128, T / 64, E), 256, 0, stream>>>(
        xbf, w1, w3, cnt, idx, slot, hbuf, 0);
    ffn_down_kernel<<<dim3(DIM / 128, T / 64, E), 256, 0, stream>>>(
        hbuf, w2, cnt, idx, wt, slot, out, 0);
}
